// GraphCompletenessPredictor_4964982194803
// MI455X (gfx1250) — matrix operand order repaired and checked by reading, not compiled, not run on hardware
//
#include <hip/hip_runtime.h>
#include <hip/hip_bf16.h>
#include <math.h>

// GraphCompletenessPredictor forward for gfx1250 (MI455X).  H = 64.

typedef __attribute__((ext_vector_type(16))) _Float16 v16h;
typedef __attribute__((ext_vector_type(8)))  float    v8f;
typedef int v4i __attribute__((vector_size(16)));
typedef __attribute__((address_space(1))) v4i* as1_v4i_ptr;
typedef __attribute__((address_space(3))) v4i* as3_v4i_ptr;

#define HDIM 64

#if defined(__HIP_DEVICE_COMPILE__) && \
    __has_builtin(__builtin_amdgcn_global_load_async_to_lds_b128) && \
    __has_builtin(__builtin_amdgcn_s_wait_asynccnt)
#define USE_ASYNC_LDS 1
#else
#define USE_ASYNC_LDS 0
#endif

static __device__ __forceinline__ v8f wmma16(v16h a, v16h b, v8f c) {
  // D = A(16x32 f16) x B(32x16 f16) + C(16x16 f32)
  return __builtin_amdgcn_wmma_f32_16x16x32_f16(
      /*neg_a=*/false, a, /*neg_b=*/false, b,
      /*c_mod=*/(short)0, c, /*reuse_a=*/false, /*reuse_b=*/false);
}

// B fragment (32x16 f16) from a row-major f32 [64][64] weight.
static __device__ __forceinline__ v16h b_frag(const float* __restrict__ W,
                                              int kt, int nt, int lane) {
  int n  = lane & 15;
  int hs = (lane >> 4) & 1;
  v16h b;
#pragma unroll
  for (int i = 0; i < 16; ++i) {
    int K = kt * 32 + hs * 16 + i;
    b[i] = (_Float16)W[(size_t)K * HDIM + nt * 16 + n];
  }
  return b;
}

// A-fragment K-local index for half i (contiguous k = 16*hs + i, matching b_frag).
static __device__ __forceinline__ int a_klocal(int i, int hs) {
  return i + 16 * hs;
}

// ---------------------------------------------------------------- utilities
__global__ void k_zero_f32(float* __restrict__ p, size_t n) {
  size_t i = (size_t)blockIdx.x * blockDim.x + threadIdx.x;
  size_t stride = (size_t)gridDim.x * blockDim.x;
  for (; i < n; i += stride) p[i] = 0.0f;
}

// ------------------------------------------------------------- degree pass
__global__ void k_degrees(const long long* __restrict__ ei,
                          float* __restrict__ deg_e,
                          float* __restrict__ deg_sl, int E) {
  int e = blockIdx.x * blockDim.x + threadIdx.x;
  if (e < E) {
    int s = (int)ei[e];
    int d = (int)ei[(size_t)E + e];
    atomicAdd(&deg_e[s], 1.0f);
    atomicAdd(&deg_e[d], 1.0f);
    atomicAdd(&deg_sl[d], 1.0f);
  }
}

// -------------------------------------------- fused edge MLP + WMMA + scatter
// ef[e] = relu(ea[e] @ We1 + be1) @ We2 + be2 ; nef[src]+=ef ; nef[dst]+=ef
// Grid-stride over 16-edge tiles so the We2 register fragments are reused.
__global__ __launch_bounds__(256, 1)
void k_edge_mlp_wmma(const float* __restrict__ ea,
                     const long long* __restrict__ ei,
                     const float* __restrict__ We1,   // [2][64]
                     const float* __restrict__ be1,   // [64]
                     const float* __restrict__ We2,   // [64][64]
                     const float* __restrict__ be2,   // [64]
                     float* __restrict__ nef,
                     int nTiles, int E) {
  __shared__ float sW[192];            // We1 row0 | We1 row1 | be1
  int lane = threadIdx.x & 31;

  for (int i = threadIdx.x; i < 192; i += blockDim.x)
    sW[i] = (i < 128) ? We1[i] : be1[i - 128];
  __syncthreads();

  // Weight fragments for We2 (registers, reused across tiles).
  v16h B[2][4];
#pragma unroll
  for (int kt = 0; kt < 2; ++kt)
#pragma unroll
    for (int nt = 0; nt < 4; ++nt) B[kt][nt] = b_frag(We2, kt, nt, lane);

  float cbias[4];
#pragma unroll
  for (int nt = 0; nt < 4; ++nt) cbias[nt] = be2[nt * 16 + (lane & 15)];

  int wave   = (blockIdx.x * blockDim.x + threadIdx.x) >> 5;
  int nwaves = (gridDim.x * blockDim.x) >> 5;
  int M  = lane & 15;
  int hs = (lane >> 4) & 1;
  int n0 = lane & 15;

  for (int t = wave; t < nTiles; t += nwaves) {
    int er = t * 16 + M;
    if (er >= E) er = E - 1;
    float ea0 = ea[(size_t)er * 2 + 0];
    float ea1 = ea[(size_t)er * 2 + 1];

    // Hidden layer-1 output (relu) -> f16 A fragments.
    v16h A[2];
#pragma unroll
    for (int kt = 0; kt < 2; ++kt) {
#pragma unroll
      for (int i = 0; i < 16; ++i) {
        int k = kt * 32 + a_klocal(i, hs);
        float tv = fmaxf(ea0 * sW[k] + ea1 * sW[64 + k] + sW[128 + k], 0.0f);
        A[kt][i] = (_Float16)tv;
      }
    }

    // src/dst for the 8 rows this lane scatters (half-tile hs).
    int rs[8], rd[8];
#pragma unroll
    for (int r = 0; r < 8; ++r) {
      int e2 = t * 16 + hs * 8 + r;
      if (e2 >= E) e2 = E - 1;
      rs[r] = (int)ei[e2];
      rd[r] = (int)ei[(size_t)E + e2];
    }

    bool full = (t * 16 + 16 <= E);
#pragma unroll
    for (int nt = 0; nt < 4; ++nt) {
      v8f c;
#pragma unroll
      for (int r = 0; r < 8; ++r) c[r] = cbias[nt];
      c = wmma16(A[0], B[0][nt], c);
      c = wmma16(A[1], B[1][nt], c);
      if (full) {
#pragma unroll
        for (int r = 0; r < 8; ++r) {
          float v = c[r];
          atomicAdd(&nef[(size_t)rs[r] * HDIM + nt * 16 + n0], v);
          atomicAdd(&nef[(size_t)rd[r] * HDIM + nt * 16 + n0], v);
        }
      } else {
#pragma unroll
        for (int r = 0; r < 8; ++r) {
          if (t * 16 + hs * 8 + r < E) {
            float v = c[r];
            atomicAdd(&nef[(size_t)rs[r] * HDIM + nt * 16 + n0], v);
            atomicAdd(&nef[(size_t)rd[r] * HDIM + nt * 16 + n0], v);
          }
        }
      }
    }
  }
}

// ------------------------------------------- node feature injection + dinv
__global__ void k_node_update(const float* __restrict__ x,
                              const float* __restrict__ nef,
                              const float* __restrict__ deg_e,
                              const float* __restrict__ deg_sl,
                              float* __restrict__ x2,
                              float* __restrict__ dinv, int N) {
  int i = blockIdx.x * blockDim.x + threadIdx.x;
  if (i < N) {
    float dg = fmaxf(deg_e[i], 1.0f);
    float inv = 0.5f / dg;
#pragma unroll
    for (int j = 0; j < 6; ++j)
      x2[(size_t)i * 6 + j] = x[(size_t)i * 6 + j] + nef[(size_t)i * HDIM + j] * inv;
    float ds = deg_sl[i] + 1.0f;              // self loop
    dinv[i] = rsqrtf(ds);
  }
}

// ------------------------------------------------------- layer 1 transform
__global__ void k_transform6(const float* __restrict__ x2,
                             const float* __restrict__ W1,   // [6][64]
                             float* __restrict__ hlin, int N) {
  int idx = blockIdx.x * blockDim.x + threadIdx.x;
  if (idx < N * HDIM) {
    int i = idx >> 6, k = idx & 63;
    float s = 0.0f;
#pragma unroll
    for (int j = 0; j < 6; ++j) s += x2[(size_t)i * 6 + j] * W1[j * HDIM + k];
    hlin[idx] = s;
  }
}

// ----------------------------------------- WMMA node transform [N,64]@[64,64]
__global__ __launch_bounds__(128, 1)
void k_transform_wmma(const float* __restrict__ hin,
                      const float* __restrict__ W,   // [64][64]
                      float* __restrict__ hlin,
                      int nTiles, int N) {
  __shared__ float tile[4][16 * HDIM];
  int lane = threadIdx.x & 31;
  int wv   = threadIdx.x >> 5;

  v16h B[2][4];
#pragma unroll
  for (int kt = 0; kt < 2; ++kt)
#pragma unroll
    for (int nt = 0; nt < 4; ++nt) B[kt][nt] = b_frag(W, kt, nt, lane);

  int nBT = (nTiles + 3) >> 2;     // block-tiles (4 tiles per block)
  for (int bt = blockIdx.x; bt < nBT; bt += gridDim.x) {
    int t = bt * 4 + wv;
    bool act = (t < nTiles);

    if (act) {
      int base = t * 16;
      bool full = (base + 16 <= N);
#if USE_ASYNC_LDS
      if (full) {
        // Async copy 16x64 f32 tile -> LDS (tracked by ASYNCcnt).
#pragma unroll 2
        for (int i = lane; i < 256; i += 32) {        // 256 x 16B
          const float* g = hin + (size_t)(base + (i >> 4)) * HDIM + (size_t)(i & 15) * 4;
          float* l = &tile[wv][(i >> 4) * HDIM + (i & 15) * 4];
          __builtin_amdgcn_global_load_async_to_lds_b128(
              (as1_v4i_ptr)(void*)g, (as3_v4i_ptr)(void*)l, 0, 0);
        }
        __builtin_amdgcn_s_wait_asynccnt(0);
      } else
#endif
      {
        float4* dstv = (float4*)tile[wv];
        const float4* srcv = (const float4*)hin;
#pragma unroll 4
        for (int i = lane; i < 256; i += 32) {
          int row = base + (i >> 4);
          float4 v = make_float4(0.f, 0.f, 0.f, 0.f);
          if (row < N) v = srcv[(size_t)row * 16 + (i & 15)];
          dstv[i] = v;
        }
        (void)full;
      }
    }
    __syncthreads();

    if (act) {
      int hs = (lane >> 4) & 1;
      int n0 = lane & 15;
      int M  = lane & 15;
      v16h A[2];
#pragma unroll
      for (int kt = 0; kt < 2; ++kt)
#pragma unroll
        for (int i = 0; i < 16; ++i)
          A[kt][i] = (_Float16)tile[wv][M * HDIM + kt * 32 + a_klocal(i, hs)];

      bool fulls = (t * 16 + 16 <= N);
#pragma unroll
      for (int nt = 0; nt < 4; ++nt) {
        v8f c = {};
        c = wmma16(A[0], B[0][nt], c);
        c = wmma16(A[1], B[1][nt], c);
        if (fulls) {
#pragma unroll
          for (int r = 0; r < 8; ++r) {
            int row = t * 16 + hs * 8 + r;
            hlin[(size_t)row * HDIM + nt * 16 + n0] = c[r];
          }
        } else {
#pragma unroll
          for (int r = 0; r < 8; ++r) {
            int row = t * 16 + hs * 8 + r;
            if (row < N) hlin[(size_t)row * HDIM + nt * 16 + n0] = c[r];
          }
        }
      }
    }
    __syncthreads();
  }
}

// ------------------------------------------------ edge aggregation (scatter)
__global__ void k_aggregate(const float* __restrict__ hlin,
                            const long long* __restrict__ ei,
                            const float* __restrict__ dinv,
                            float* __restrict__ agg, int E) {
  int lane = threadIdx.x & 31;
  int wave = (blockIdx.x * blockDim.x + threadIdx.x) >> 5;
  int nwaves = (gridDim.x * blockDim.x) >> 5;
  for (int e = wave; e < E; e += nwaves) {
    // Prefetch next iteration's edge indices (global_prefetch_b8).
    __builtin_prefetch(&ei[(size_t)e + nwaves], 0, 1);
    __builtin_prefetch(&ei[(size_t)E + e + nwaves], 0, 1);
    int s = (int)ei[e];
    int d = (int)ei[(size_t)E + e];
    float nrm = dinv[s] * dinv[d];
    float v0 = hlin[(size_t)s * HDIM + lane];
    float v1 = hlin[(size_t)s * HDIM + 32 + lane];
    atomicAdd(&agg[(size_t)d * HDIM + lane],      nrm * v0);
    atomicAdd(&agg[(size_t)d * HDIM + 32 + lane], nrm * v1);
  }
}

// ------------------------------------ self-loop + bias (+relu) epilogue
__global__ void k_finish(const float* __restrict__ agg,
                         const float* __restrict__ hlin,
                         const float* __restrict__ dinv,
                         const float* __restrict__ b,
                         float* __restrict__ hout, int N, int do_relu) {
  int idx = blockIdx.x * blockDim.x + threadIdx.x;
  if (idx < N * HDIM) {
    int i = idx >> 6, k = idx & 63;
    float di = dinv[i];
    float v = agg[idx] + di * di * hlin[idx] + b[k];
    if (do_relu) v = fmaxf(v, 0.0f);
    hout[idx] = v;
  }
}

// ----------------------------------------------- column sum / sum-of-squares
__global__ void k_colstats(const float* __restrict__ h,
                           float* __restrict__ gsum,
                           float* __restrict__ gsumsq, int N) {
  __shared__ float s1[256], s2[256];
  int k = threadIdx.x & 63;
  int g = threadIdx.x >> 6;
  float a = 0.0f, b = 0.0f;
  for (int i = blockIdx.x * 4 + g; i < N; i += gridDim.x * 4) {
    float v = h[(size_t)i * HDIM + k];
    a += v; b += v * v;
  }
  s1[threadIdx.x] = a; s2[threadIdx.x] = b;
  __syncthreads();
  if (threadIdx.x < 64) {
    float ta = s1[k] + s1[64 + k] + s1[128 + k] + s1[192 + k];
    float tb = s2[k] + s2[64 + k] + s2[128 + k] + s2[192 + k];
    atomicAdd(&gsum[k], ta);
    atomicAdd(&gsumsq[k], tb);
  }
}

// ----------------------------------------------------- stats + MLP head
__global__ void k_head(const float* __restrict__ gsum,
                       const float* __restrict__ gsumsq,
                       const float* __restrict__ Wp1, const float* __restrict__ bp1,
                       const float* __restrict__ Wp2, const float* __restrict__ bp2,
                       const float* __restrict__ Wp3, const float* __restrict__ bp3,
                       float* __restrict__ out, int N) {
  __shared__ float comb[192], p1[64], p2[32];
  int t = threadIdx.x;
  if (t < 64) {
    float mean = gsum[t] / (float)N;
    float var  = (gsumsq[t] - (float)N * mean * mean) / (float)(N - 1);
    float sd   = sqrtf(fmaxf(var, 0.0f));
    comb[t] = mean;          // graph_emb (mean pool)
    comb[64 + t] = mean;     // node_mean
    comb[128 + t] = sd;      // node_std (unbiased)
  }
  __syncthreads();
  if (t < 64) {
    float s = bp1[t];
    for (int j = 0; j < 192; ++j) s += comb[j] * Wp1[j * 64 + t];
    p1[t] = fmaxf(s, 0.0f);
  }
  __syncthreads();
  if (t < 32) {
    float s = bp2[t];
    for (int j = 0; j < 64; ++j) s += p1[j] * Wp2[j * 32 + t];
    p2[t] = fmaxf(s, 0.0f);
  }
  __syncthreads();
  if (t == 0) {
    float s = bp3[0];
    for (int j = 0; j < 32; ++j) s += p2[j] * Wp3[j];
    out[0] = 1.0f / (1.0f + expf(-s));
  }
}

// ======================================================================
extern "C" void kernel_launch(void* const* d_in, const int* in_sizes, int n_in,
                              void* d_out, int out_size, void* d_ws, size_t ws_size,
                              hipStream_t stream) {
  const float*     x    = (const float*)d_in[0];
  const long long* ei   = (const long long*)d_in[1];
  const float*     ea   = (const float*)d_in[2];
  const float *W1 = (const float*)d_in[3],  *b1  = (const float*)d_in[4];
  const float *W2 = (const float*)d_in[5],  *b2  = (const float*)d_in[6];
  const float *W3 = (const float*)d_in[7],  *b3  = (const float*)d_in[8];
  const float *We1= (const float*)d_in[9],  *be1 = (const float*)d_in[10];
  const float *We2= (const float*)d_in[11], *be2 = (const float*)d_in[12];
  const float *Wp1= (const float*)d_in[13], *bp1 = (const float*)d_in[14];
  const float *Wp2= (const float*)d_in[15], *bp2 = (const float*)d_in[16];
  const float *Wp3= (const float*)d_in[17], *bp3 = (const float*)d_in[18];
  float* out = (float*)d_out;

  const int N = in_sizes[0] / 6;
  const int E = in_sizes[2] / 2;

  // Workspace layout (floats).
  float* ws     = (float*)d_ws;
  float* nef    = ws;                          // [N,64]  (also reused as AGG)
  float* P      = nef    + (size_t)N * HDIM;   // [N,64]  h_lin
  float* Q      = P      + (size_t)N * HDIM;   // [N,64]  h (layer output)
  float* deg_e  = Q      + (size_t)N * HDIM;   // [N]
  float* deg_sl = deg_e  + (size_t)N;          // [N]
  float* dinv   = deg_sl + (size_t)N;          // [N]
  float* x2     = dinv   + (size_t)N;          // [N,6]
  float* gsum   = x2     + (size_t)N * 6;      // [64]
  float* gsumsq = gsum   + 64;                 // [64]

  const size_t nh = (size_t)N * HDIM;
  const int ZB = 2048;

  // --- zero: nef, degrees, accumulators
  k_zero_f32<<<ZB, 256, 0, stream>>>(nef, nh);
  k_zero_f32<<<64, 256, 0, stream>>>(deg_e, (size_t)2 * N);
  k_zero_f32<<<1, 128, 0, stream>>>(gsum, 128);

  // --- degrees
  k_degrees<<<(E + 255) / 256, 256, 0, stream>>>(ei, deg_e, deg_sl, E);

  // --- fused edge MLP (WMMA) scattering into nef
  int eTiles = (E + 15) / 16;
  k_edge_mlp_wmma<<<1250, 256, 0, stream>>>(ea, ei, We1, be1, We2, be2, nef,
                                            eTiles, E);

  // --- node feature injection + GCN dinv
  k_node_update<<<(N + 255) / 256, 256, 0, stream>>>(x, nef, deg_e, deg_sl,
                                                     x2, dinv, N);

  const int gridNH = (int)((nh + 255) / 256);
  int nTiles = (N + 15) / 16;
  int gridT  = 512;

  // ---------- GCN layer 1 (K=6 transform, scalar)
  k_zero_f32<<<ZB, 256, 0, stream>>>(nef, nh);                       // AGG
  k_transform6<<<gridNH, 256, 0, stream>>>(x2, W1, P, N);
  k_aggregate<<<4096, 256, 0, stream>>>(P, ei, dinv, nef, E);
  k_finish<<<gridNH, 256, 0, stream>>>(nef, P, dinv, b1, Q, N, 1);

  // ---------- GCN layer 2 (WMMA transform)
  k_zero_f32<<<ZB, 256, 0, stream>>>(nef, nh);
  k_transform_wmma<<<gridT, 128, 0, stream>>>(Q, W2, P, nTiles, N);
  k_aggregate<<<4096, 256, 0, stream>>>(P, ei, dinv, nef, E);
  k_finish<<<gridNH, 256, 0, stream>>>(nef, P, dinv, b2, Q, N, 1);

  // ---------- GCN layer 3 (WMMA transform, no relu)
  k_zero_f32<<<ZB, 256, 0, stream>>>(nef, nh);
  k_transform_wmma<<<gridT, 128, 0, stream>>>(Q, W3, P, nTiles, N);
  k_aggregate<<<4096, 256, 0, stream>>>(P, ei, dinv, nef, E);
  k_finish<<<gridNH, 256, 0, stream>>>(nef, P, dinv, b3, Q, N, 0);

  // ---------- pooling stats + predictor head
  k_colstats<<<512, 256, 0, stream>>>(Q, gsum, gsumsq, N);
  k_head<<<1, 64, 0, stream>>>(gsum, gsumsq, Wp1, bp1, Wp2, bp2, Wp3, bp3,
                               out, N);
}
